// AttentionLSTM_60292750901484
// MI455X (gfx1250) — compile-verified, matrix operand order checked by reading, not run on hardware
//
#include <hip/hip_runtime.h>
#include <math.h>

// ---------------- problem dims (fixed by reference) ----------------
#define N_B   256
#define T_S   32
#define D_IN  512
#define H_DIM 512
#define C_CH  1280
#define P2    49     // 7*7 spatial
#define G4H   2048   // 4*H

#define LDSTRIDE 40  // bf16 elems per LDS tile line (32 data + 8 pad; 80B keeps 16B align)

typedef __bf16 bf16_t;
typedef __attribute__((ext_vector_type(16))) __bf16 v16bf;
typedef __attribute__((ext_vector_type(8)))  __bf16 v8bf;
typedef __attribute__((ext_vector_type(4)))  __bf16 v4bf;
typedef __attribute__((ext_vector_type(2)))  __bf16 v2bf;
typedef __attribute__((ext_vector_type(8)))  float  v8f;

// native f32 -> bf16 (hardware RNE convert, lets clang use v_cvt_*_bf16_f32)
__device__ __forceinline__ bf16_t f2bf(float f) { return (bf16_t)f; }

__device__ __forceinline__ float sigmoid_f(float x) {
  return 1.0f / (1.0f + expf(-x));
}

// ---- LDS staging: tiles stored fragment-native as [line 64][k 32] bf16 ----

// A-tile: 64 rows x 32 k from a row-major matrix (k contiguous).
// float4 global loads; one packed 8-byte LDS store per 4 elements.
__device__ __forceinline__ void stage_A(const float* __restrict__ src, long ldRow,
                                        int rowBase, int kb, bf16_t* __restrict__ dst,
                                        int tid) {
#pragma unroll
  for (int it = 0; it < 4; ++it) {
    int g   = tid + it * 128;   // float4 group 0..511
    int row = g >> 3;           // 0..63
    int ko  = (g & 7) * 4;      // 0..28
    const float4 v = *(const float4*)(src + (long)(rowBase + row) * ldRow + kb + ko);
    v4bf pk;
    pk[0] = f2bf(v.x); pk[1] = f2bf(v.y); pk[2] = f2bf(v.z); pk[3] = f2bf(v.w);
    *(v4bf*)(dst + row * LDSTRIDE + ko) = pk;   // 8B aligned (80B lines, ko even*2)
  }
}

// B-tile: 32 k x 64 cols from a row-major K x Ncols matrix (cols contiguous).
// Two k-rows per thread-group so each col stores a k-pair as one 4B LDS store.
__device__ __forceinline__ void stage_B(const float* __restrict__ src, long ldRow,
                                        int colBase, int kb, bf16_t* __restrict__ dst,
                                        int tid) {
#pragma unroll
  for (int it = 0; it < 2; ++it) {
    int g  = tid + it * 128;    // 0..255
    int k  = (g >> 4) * 2;      // 0,2,..,30
    int c4 = (g & 15) * 4;      // 0..60
    const float* p0 = src + (long)(kb + k) * ldRow + colBase + c4;
    const float4 r0 = *(const float4*)(p0);
    const float4 r1 = *(const float4*)(p0 + ldRow);
    v2bf q0; q0[0] = f2bf(r0.x); q0[1] = f2bf(r1.x);
    v2bf q1; q1[0] = f2bf(r0.y); q1[1] = f2bf(r1.y);
    v2bf q2; q2[0] = f2bf(r0.z); q2[1] = f2bf(r1.z);
    v2bf q3; q3[0] = f2bf(r0.w); q3[1] = f2bf(r1.w);
    *(v2bf*)(dst + (c4 + 0) * LDSTRIDE + k) = q0;  // 4B aligned (k even)
    *(v2bf*)(dst + (c4 + 1) * LDSTRIDE + k) = q1;
    *(v2bf*)(dst + (c4 + 2) * LDSTRIDE + k) = q2;
    *(v2bf*)(dst + (c4 + 3) * LDSTRIDE + k) = q3;
  }
}

// Conv B-tile: B[k][p] = An[k*49+p], zero-padded p>=49. Branchless clamp+select;
// packs k-pairs into one 4B LDS store.
__device__ __forceinline__ void stage_B_conv(const float* __restrict__ An, int kb,
                                             bf16_t* __restrict__ dst, int tid) {
#pragma unroll
  for (int it = 0; it < 8; ++it) {
    int e = tid + it * 128;     // 0..1023
    int k = (e >> 6) * 2;       // 0,2,..,30
    int p = e & 63;             // 0..63
    int pc = (p < P2) ? p : 0;
    const float* q = An + (long)(kb + k) * P2 + pc;
    float v0 = q[0];
    float v1 = q[P2];
    v0 = (p < P2) ? v0 : 0.0f;
    v1 = (p < P2) ? v1 : 0.0f;
    v2bf pk; pk[0] = f2bf(v0); pk[1] = f2bf(v1);
    *(v2bf*)(dst + p * LDSTRIDE + k) = pk;
  }
}

// Fragment from LDS tile: lane l reads line rc, K halves k0 / k0+16 -> two ds_load_b128.
__device__ __forceinline__ v16bf frag_from_lds(const bf16_t* __restrict__ tile,
                                               int rc, int lane) {
  int k0 = ((lane >> 4) & 1) * 8;
  const bf16_t* p = tile + rc * LDSTRIDE + k0;
  v8bf lo = *(const v8bf*)(p);
  v8bf hi = *(const v8bf*)(p + 16);
  return __builtin_shufflevector(lo, hi, 0, 1, 2, 3, 4, 5, 6, 7,
                                 8, 9, 10, 11, 12, 13, 14, 15);
}

// ---------------- 1) conv 1x1 projection ----------------
// A_flat[n,h,p] = sum_c Wconv[h,c]*A[n,c,p] + bconv[h]
// 128 threads (4 waves) -> 64 h-rows x 64 p-cols (49 valid), K=1280, double-buffered.
__global__ __launch_bounds__(128)
void conv_proj_kernel(const float* __restrict__ A, const float* __restrict__ Wconv,
                      const float* __restrict__ bconv, float* __restrict__ Aflat) {
  __shared__ alignas(16) bf16_t sA[2][64 * LDSTRIDE];
  __shared__ alignas(16) bf16_t sB[2][64 * LDSTRIDE];
  const int tid  = threadIdx.x;
  const int lane = tid & 31;
  const int wave = tid >> 5;
  const int rowBase = blockIdx.x * 64;   // h tile: 0..7
  const int n       = blockIdx.y;        // batch
  const float* An = A + (long)n * C_CH * P2;

  v8f acc0 = {}, acc1 = {}, acc2 = {}, acc3 = {};
  const int arow = 16 * wave + (lane & 15);
  const int brow = lane & 15;
  const int NCH = C_CH / 32;             // 40 chunks

  stage_A(Wconv, C_CH, rowBase, 0, sA[0], tid);
  stage_B_conv(An, 0, sB[0], tid);
  __syncthreads();

#pragma unroll 1
  for (int i = 0; i < NCH; ++i) {
    const int cur = i & 1;
    if (i + 1 < NCH) {                   // prefetch next chunk into idle buffer
      stage_A(Wconv, C_CH, rowBase, (i + 1) * 32, sA[cur ^ 1], tid);
      stage_B_conv(An, (i + 1) * 32, sB[cur ^ 1], tid);
    }
    v16bf af = frag_from_lds(sA[cur], arow, lane);
    v16bf b0 = frag_from_lds(sB[cur], brow, lane);
    v16bf b1 = frag_from_lds(sB[cur], 16 + brow, lane);
    v16bf b2 = frag_from_lds(sB[cur], 32 + brow, lane);
    v16bf b3 = frag_from_lds(sB[cur], 48 + brow, lane);
    acc0 = __builtin_amdgcn_wmma_f32_16x16x32_bf16(false, af, false, b0, (short)0, acc0, false, false);
    acc1 = __builtin_amdgcn_wmma_f32_16x16x32_bf16(false, af, false, b1, (short)0, acc1, false, false);
    acc2 = __builtin_amdgcn_wmma_f32_16x16x32_bf16(false, af, false, b2, (short)0, acc2, false, false);
    acc3 = __builtin_amdgcn_wmma_f32_16x16x32_bf16(false, af, false, b3, (short)0, acc3, false, false);
    __syncthreads();
  }

  const int pj     = lane & 15;
  const int rowOff = rowBase + 16 * wave + ((lane >> 4) & 1) * 8;
#pragma unroll
  for (int r = 0; r < 8; ++r) {
    int h = rowOff + r;
    float bb = bconv[h];
    float* dst = Aflat + ((long)n * H_DIM + h) * P2;
    if (pj      < P2) dst[pj]      = acc0[r] + bb;
    if (16 + pj < P2) dst[16 + pj] = acc1[r] + bb;
    if (32 + pj < P2) dst[32 + pj] = acc2[r] + bb;
    if (48 + pj < P2) dst[48 + pj] = acc3[r] + bb;
  }
}

// ---------------- 2) h0 = c0 = mean over spatial of A_flat ----------------
__global__ __launch_bounds__(256)
void init_h0_kernel(const float* __restrict__ Aflat, float* __restrict__ h,
                    float* __restrict__ c) {
  const int n = blockIdx.x;
  for (int hh = threadIdx.x; hh < H_DIM; hh += blockDim.x) {
    const float* row = Aflat + ((long)n * H_DIM + hh) * P2;
    float s = 0.0f;
#pragma unroll
    for (int q = 0; q < P2; ++q) s += row[q];
    s *= (1.0f / 49.0f);
    h[(long)n * H_DIM + hh] = s;
    c[(long)n * H_DIM + hh] = s;
  }
}

// ---------------- 3a) attention (one WG per batch row) ----------------
__global__ __launch_bounds__(128)
void attn_kernel(const float* __restrict__ Aflat, const float* __restrict__ h,
                 float* __restrict__ attn) {
  __shared__ float s_h[H_DIM];
  __shared__ float s_w[P2];
  const int n = blockIdx.x, tid = threadIdx.x;

  for (int i = tid; i < H_DIM; i += 128) s_h[i] = h[(long)n * H_DIM + i];
  __syncthreads();

  const float* An = Aflat + (long)n * H_DIM * P2;
  if (tid < P2) {
    float s = 0.0f;
    for (int hh = 0; hh < H_DIM; ++hh) s += s_h[hh] * An[(long)hh * P2 + tid];
    s_w[tid] = s * 0.04419417382415922f;  // 1/sqrt(512)
  }
  __syncthreads();
  if (tid == 0) {  // 49-wide softmax: serial is cheap
    float m = s_w[0];
    for (int q = 1; q < P2; ++q) m = fmaxf(m, s_w[q]);
    float sum = 0.0f;
    for (int q = 0; q < P2; ++q) { float e = expf(s_w[q] - m); s_w[q] = e; sum += e; }
    float inv = 1.0f / sum;
    for (int q = 0; q < P2; ++q) s_w[q] *= inv;
  }
  __syncthreads();
  for (int hh = tid; hh < H_DIM; hh += 128) {
    const float* row = An + (long)hh * P2;
    float s = 0.0f;
#pragma unroll
    for (int q = 0; q < P2; ++q) s += row[q] * s_w[q];
    attn[(long)n * H_DIM + hh] = s;
  }
}

// ---------------- 3b) fused gate preactivations ----------------
// a[n,j] = xt·Wx + h·Wh + attn·Wattn + b    (K fused = 1536, 48 chunks)
// 128 threads (4 waves) -> 64 n-rows x 64 j-cols, double-buffered LDS.
__global__ __launch_bounds__(128)
void lstm_gemm_kernel(const float* __restrict__ x, int t,
                      const float* __restrict__ h, const float* __restrict__ attn,
                      const float* __restrict__ Wx, const float* __restrict__ Wh,
                      const float* __restrict__ Wattn, const float* __restrict__ bias,
                      float* __restrict__ a_out) {
  __shared__ alignas(16) bf16_t sA[2][64 * LDSTRIDE];
  __shared__ alignas(16) bf16_t sB[2][64 * LDSTRIDE];
  const int tid  = threadIdx.x;
  const int lane = tid & 31;
  const int wave = tid >> 5;
  const int colBase = blockIdx.x * 64;   // j tile: 0..31
  const int rowBase = blockIdx.y * 64;   // n tile: 0..3

  const float* Asrc[3] = { x + (long)t * D_IN, h, attn };
  const long   Ald[3]  = { (long)T_S * D_IN, H_DIM, H_DIM };
  const float* Bsrc[3] = { Wx, Wh, Wattn };

  v8f acc0 = {}, acc1 = {}, acc2 = {}, acc3 = {};
  const int arow = 16 * wave + (lane & 15);
  const int brow = lane & 15;
  const int NCH = 48;                    // 3 segments x 16 chunks

  auto stage_chunk = [&](int i, bf16_t* bufA, bf16_t* bufB) {
    int s  = i >> 4;
    int kb = (i & 15) << 5;
    stage_A(Asrc[s], Ald[s], rowBase, kb, bufA, tid);
    stage_B(Bsrc[s], G4H, colBase, kb, bufB, tid);
  };

  stage_chunk(0, sA[0], sB[0]);
  __syncthreads();

#pragma unroll 1
  for (int i = 0; i < NCH; ++i) {
    const int cur = i & 1;
    if (i + 1 < NCH) stage_chunk(i + 1, sA[cur ^ 1], sB[cur ^ 1]);
    v16bf af = frag_from_lds(sA[cur], arow, lane);
    v16bf b0 = frag_from_lds(sB[cur], brow, lane);
    v16bf b1 = frag_from_lds(sB[cur], 16 + brow, lane);
    v16bf b2 = frag_from_lds(sB[cur], 32 + brow, lane);
    v16bf b3 = frag_from_lds(sB[cur], 48 + brow, lane);
    acc0 = __builtin_amdgcn_wmma_f32_16x16x32_bf16(false, af, false, b0, (short)0, acc0, false, false);
    acc1 = __builtin_amdgcn_wmma_f32_16x16x32_bf16(false, af, false, b1, (short)0, acc1, false, false);
    acc2 = __builtin_amdgcn_wmma_f32_16x16x32_bf16(false, af, false, b2, (short)0, acc2, false, false);
    acc3 = __builtin_amdgcn_wmma_f32_16x16x32_bf16(false, af, false, b3, (short)0, acc3, false, false);
    __syncthreads();
  }

  const int jn     = lane & 15;
  const int rowOff = rowBase + 16 * wave + ((lane >> 4) & 1) * 8;
#pragma unroll
  for (int r = 0; r < 8; ++r) {
    int n = rowOff + r;
    float* dst = a_out + (long)n * G4H + colBase;
    dst[jn]      = acc0[r] + bias[colBase + jn];
    dst[16 + jn] = acc1[r] + bias[colBase + 16 + jn];
    dst[32 + jn] = acc2[r] + bias[colBase + 32 + jn];
    dst[48 + jn] = acc3[r] + bias[colBase + 48 + jn];
  }
}

// ---------------- 3c) gates, state update, write hn[n,t,:] ----------------
__global__ __launch_bounds__(256)
void lstm_gate_kernel(const float* __restrict__ a, int t,
                      float* __restrict__ c, float* __restrict__ h,
                      float* __restrict__ out) {
  const int n = blockIdx.x;
  const float* an = a + (long)n * G4H;
  for (int hh = threadIdx.x; hh < H_DIM; hh += blockDim.x) {
    float iv = sigmoid_f(an[hh]);
    float fv = sigmoid_f(an[H_DIM + hh]);
    float ov = sigmoid_f(an[2 * H_DIM + hh]);
    float gv = tanhf(an[3 * H_DIM + hh]);
    long idx = (long)n * H_DIM + hh;
    float c2 = fv * c[idx] + iv * gv;
    float h2 = ov * tanhf(c2);
    c[idx] = c2;
    h[idx] = h2;
    out[((long)n * T_S + t) * H_DIM + hh] = h2;
  }
}

extern "C" void kernel_launch(void* const* d_in, const int* in_sizes, int n_in,
                              void* d_out, int out_size, void* d_ws, size_t ws_size,
                              hipStream_t stream) {
  (void)in_sizes; (void)n_in; (void)out_size; (void)ws_size;
  const float* x     = (const float*)d_in[0];  // [N,T,D]
  const float* A     = (const float*)d_in[1];  // [N,C,7,7]
  const float* Wx    = (const float*)d_in[2];  // [D,4H]
  const float* Wh    = (const float*)d_in[3];  // [H,4H]
  const float* Wattn = (const float*)d_in[4];  // [H,4H]
  const float* b     = (const float*)d_in[5];  // [4H]
  const float* Wconv = (const float*)d_in[6];  // [H,C]
  const float* bconv = (const float*)d_in[7];  // [H]
  float* out = (float*)d_out;                  // [N,T,H]

  // workspace carve-up (floats): A_flat | h | c | attn | preact   (~30 MB)
  float* ws    = (float*)d_ws;
  float* Aflat = ws;
  float* hbuf  = Aflat + (size_t)N_B * H_DIM * P2;
  float* cbuf  = hbuf  + (size_t)N_B * H_DIM;
  float* attn  = cbuf  + (size_t)N_B * H_DIM;
  float* abuf  = attn  + (size_t)N_B * H_DIM;

  // 1) conv projection (big parallel WMMA GEMM) + bias
  conv_proj_kernel<<<dim3(H_DIM / 64, N_B), 128, 0, stream>>>(A, Wconv, bconv, Aflat);
  // 2) h0 = c0 = spatial mean
  init_h0_kernel<<<N_B, 256, 0, stream>>>(Aflat, hbuf, cbuf);
  // 3) sequential recurrence
  for (int t = 0; t < T_S; ++t) {
    attn_kernel<<<N_B, 128, 0, stream>>>(Aflat, hbuf, attn);
    lstm_gemm_kernel<<<dim3(G4H / 64, N_B / 64), 128, 0, stream>>>(
        x, t, hbuf, attn, Wx, Wh, Wattn, b, abuf);
    lstm_gate_kernel<<<N_B, 256, 0, stream>>>(abuf, t, cbuf, hbuf, out);
  }
}
